// GAT_63359357550604
// MI455X (gfx1250) — matrix operand order repaired and checked by reading, not compiled, not run on hardware
//
#include <hip/hip_runtime.h>

// ---------------------------------------------------------------------------
// CDNA5 (gfx1250) GAT inference: bf16 WMMA GEMMs + fp32 edge softmax/scatter.
// ---------------------------------------------------------------------------

typedef __attribute__((ext_vector_type(16))) __bf16 v16bf;
typedef __attribute__((ext_vector_type(8)))  __bf16 v8bf;
typedef __attribute__((ext_vector_type(2)))  __bf16 v2bf;
typedef __attribute__((ext_vector_type(8)))  float  v8f;

#define NEG_SLOPE 0.2f
#define WPB 8   // waves per GEMM block (fixed: 256 threads)

// Monotone float <-> uint encoding so atomicMax(u32) == float max.
__device__ __forceinline__ unsigned fenc(float f) {
    unsigned u = __float_as_uint(f);
    return (u & 0x80000000u) ? ~u : (u | 0x80000000u);
}
__device__ __forceinline__ float fdec(unsigned u) {
    unsigned v = (u & 0x80000000u) ? (u & 0x7FFFFFFFu) : ~u;
    return __uint_as_float(v);
}

// ---------------------------------------------------------------------------
// Pack W[K x Nlog] (row-major f32) into per-lane WMMA B-fragment order (bf16).
// Fragment layout (this code's contiguous k order, not the ISA table order): lane l holds column n = l&15,
// K-range half*16 .. half*16+15 (half = l>>4), elements contiguous in K.
// Storage: [coltile][kstep][lane][16 bf16] fully contiguous -> 32B per lane.
// ---------------------------------------------------------------------------
__global__ void pack_B_kernel(const float* __restrict__ W, unsigned short* __restrict__ BpRaw,
                              int K, int Nlog, int coltiles) {
    int idx = blockIdx.x * blockDim.x + threadIdx.x;
    int ksteps = K >> 5;
    int total = coltiles * ksteps * 32;
    if (idx >= total) return;
    __bf16* Bp = (__bf16*)BpRaw;
    int lane = idx & 31;
    int rest = idx >> 5;
    int ks = rest % ksteps;
    int ct = rest / ksteps;
    int half = lane >> 4;
    int n = ct * 16 + (lane & 15);
    int kbase = ks * 32 + half * 16;
    __bf16* dst = Bp + (size_t)idx * 16;
#pragma unroll
    for (int i = 0; i < 16; ++i) {
        float v = (n < Nlog) ? W[(size_t)(kbase + i) * Nlog + n] : 0.0f;
        dst[i] = (__bf16)v;
    }
}

// ---------------------------------------------------------------------------
// C[M x Nlog] = A[M x K] * B (packed). 8 waves/block share a 16x32 bf16
// A-tile in LDS; each wave owns TWO 16x16 output tiles (2 accumulators,
// 2 WMMAs per staged A-tile). Staging is straight-line: one float2 global
// load + one packed bf16 LDS store per thread per k-step.
// ---------------------------------------------------------------------------
__global__ __launch_bounds__(256)
void wmma_gemm_kernel(const float* __restrict__ A,
                      const unsigned short* __restrict__ BpRaw,
                      float* __restrict__ C,
                      int M, int K, int Nlog, int coltiles) {
    __shared__ __attribute__((aligned(16))) __bf16 sA[16][32];
    const __bf16* Bp = (const __bf16*)BpRaw;

    const int lane = threadIdx.x;          // 0..31
    const int wid  = threadIdx.y;          // 0..7
    const int tid  = wid * 32 + lane;

    const int rowbase = blockIdx.y * 16;
    const int ct0 = (blockIdx.x * WPB + wid) * 2;
    const int ct1 = ct0 + 1;
    const bool act0 = (ct0 < coltiles);    // wave-uniform
    const bool act1 = (ct1 < coltiles);    // wave-uniform

    const int m    = lane & 15;
    const int half = lane >> 4;
    const int ksteps = K >> 5;

    // Staging assignment: thread stages 2 consecutive floats of the A tile.
    const int sr = tid >> 4;               // row 0..15
    const int sk = (tid & 15) * 2;         // col 0,2,..,30
    const int srow = rowbase + sr;
    const float* aRow = A + (size_t)srow * K + sk;

    const __bf16* bPtr0 = Bp + ((size_t)ct0 * ksteps * 32 + lane) * 16;
    const __bf16* bPtr1 = Bp + ((size_t)ct1 * ksteps * 32 + lane) * 16;
    const size_t bStep = (size_t)32 * 16;  // elements per k-step per coltile

    v8f acc0 = {}, acc1 = {};

    for (int ks = 0; ks < ksteps; ++ks) {
        __syncthreads();
        {
            float2 v = make_float2(0.f, 0.f);
            if (srow < M) v = *(const float2*)(aRow + ks * 32);
            v2bf p;
            p[0] = (__bf16)v.x;
            p[1] = (__bf16)v.y;
            *(v2bf*)&sA[sr][sk] = p;       // packed ds_store_b32
        }
        __syncthreads();

        // A fragment (ISA 7.12.2, 16-bit A 16x32): two 16B chunks per lane.
        v8bf lo = *(const v8bf*)&sA[m][half * 16];
        v8bf hi = *(const v8bf*)&sA[m][half * 16 + 8];
        v16bf a;
#pragma unroll
        for (int i = 0; i < 8; ++i) { a[i] = lo[i]; a[i + 8] = hi[i]; }

        if (act0) {
            v16bf b = *(const v16bf*)(bPtr0 + (size_t)ks * bStep);
            acc0 = __builtin_amdgcn_wmma_f32_16x16x32_bf16(
                false, a, false, b, (short)0, acc0, false, false);
        }
        if (act1) {
            v16bf b = *(const v16bf*)(bPtr1 + (size_t)ks * bStep);
            acc1 = __builtin_amdgcn_wmma_f32_16x16x32_bf16(
                false, a, false, b, (short)0, acc1, false, false);
        }
    }

    // C/D layout: VGPR r -> row rowbase + half*8 + r, col = ct*16 + (lane&15).
    if (act0) {
        int col = ct0 * 16 + m;
        if (col < Nlog) {
#pragma unroll
            for (int r = 0; r < 8; ++r) {
                int row = rowbase + half * 8 + r;
                if (row < M) C[(size_t)row * Nlog + col] = acc0[r];
            }
        }
    }
    if (act1) {
        int col = ct1 * 16 + m;
        if (col < Nlog) {
#pragma unroll
            for (int r = 0; r < 8; ++r) {
                int row = rowbase + half * 8 + r;
                if (row < M) C[(size_t)row * Nlog + col] = acc1[r];
            }
        }
    }
}

// ---------------------------------------------------------------------------
// a_src[n,h] = <H[n,h,:], att_src[h,:]>, same for a_dst.
// ---------------------------------------------------------------------------
__global__ void att_dots_kernel(const float* __restrict__ H,
                                const float* __restrict__ attS,
                                const float* __restrict__ attD,
                                float* __restrict__ aS, float* __restrict__ aD,
                                int Nn, int heads, int C, int rs) {
    int i = blockIdx.x * blockDim.x + threadIdx.x;
    if (i >= Nn * heads) return;
    int n = i / heads, h = i % heads;
    const float* hp = H + (size_t)n * rs + h * C;
    const float* sp = attS + h * C;
    const float* dp = attD + h * C;
    float s = 0.f, d = 0.f;
    for (int c = 0; c < C; ++c) {
        float v = hp[c];
        s += v * sp[c];
        d += v * dp[c];
    }
    aS[i] = s;
    aD[i] = d;
}

// ---------------------------------------------------------------------------
// Edge pass 1: segment max of leaky_relu(a_src[s]+a_dst[d]) via ordered-uint
// atomicMax. emaxU zero-initialized; every node has a self-loop so every
// segment receives at least one real value.
// ---------------------------------------------------------------------------
__global__ void edge_max_kernel(const int* __restrict__ src, const int* __restrict__ dst,
                                int E, int Nn,
                                const float* __restrict__ aS, const float* __restrict__ aD,
                                unsigned* __restrict__ emaxU, int heads) {
    int e = blockIdx.x * blockDim.x + threadIdx.x;
    int EP = E + Nn;
    if (e >= EP) return;
    int s, d;
    if (e < E) { s = src[e]; d = dst[e]; } else { s = d = e - E; }
    for (int h = 0; h < heads; ++h) {
        float v = aS[(size_t)s * heads + h] + aD[(size_t)d * heads + h];
        v = v > 0.f ? v : NEG_SLOPE * v;
        atomicMax(&emaxU[(size_t)d * heads + h], fenc(v));
    }
}

// ---------------------------------------------------------------------------
// Edge pass 2: ex = exp(e - max[dst]); store per-edge, atomicAdd denominators.
// ---------------------------------------------------------------------------
__global__ void edge_expsum_kernel(const int* __restrict__ src, const int* __restrict__ dst,
                                   int E, int Nn,
                                   const float* __restrict__ aS, const float* __restrict__ aD,
                                   const unsigned* __restrict__ emaxU,
                                   float* __restrict__ exw, float* __restrict__ denom,
                                   int heads) {
    int e = blockIdx.x * blockDim.x + threadIdx.x;
    int EP = E + Nn;
    if (e >= EP) return;
    int s, d;
    if (e < E) { s = src[e]; d = dst[e]; } else { s = d = e - E; }
    for (int h = 0; h < heads; ++h) {
        float v = aS[(size_t)s * heads + h] + aD[(size_t)d * heads + h];
        v = v > 0.f ? v : NEG_SLOPE * v;
        float ex = __expf(v - fdec(emaxU[(size_t)d * heads + h]));
        exw[(size_t)e * heads + h] = ex;
        atomicAdd(&denom[(size_t)d * heads + h], ex);
    }
}

// ---------------------------------------------------------------------------
// Edge pass 3: out[d,:] += alpha[e,h(c)] * H[s,:]. One wave per edge; each
// lane handles 4 consecutive channels (float4 gather = global_load_b128,
// coalesced 512B/wave) then 4 f32 atomics. c0 % 4 == 0 and heads split at 64
// -> all 4 channels share one head (h = c0 >> 6; layer 2: heads=1, c0 < 40).
// ---------------------------------------------------------------------------
__global__ void edge_scatter_kernel(const int* __restrict__ src, const int* __restrict__ dst,
                                    int E, int Nn,
                                    const float* __restrict__ H,
                                    const float* __restrict__ exw,
                                    const float* __restrict__ denom,
                                    float* __restrict__ out,
                                    int heads, int rs) {
    int lane = threadIdx.x & 31;
    int e = (blockIdx.x * blockDim.x + threadIdx.x) >> 5;
    int EP = E + Nn;
    if (e >= EP) return;
    int s, d;
    if (e < E) { s = src[e]; d = dst[e]; } else { s = d = e - E; }
    const float* hrow = H + (size_t)s * rs;
    float* orow = out + (size_t)d * rs;
    for (int c0 = lane * 4; c0 < rs; c0 += 128) {
        float4 hv = *(const float4*)(hrow + c0);
        int h = c0 >> 6;
        float alpha = exw[(size_t)e * heads + h] / denom[(size_t)d * heads + h];
        atomicAdd(&orow[c0 + 0], alpha * hv.x);
        atomicAdd(&orow[c0 + 1], alpha * hv.y);
        atomicAdd(&orow[c0 + 2], alpha * hv.z);
        atomicAdd(&orow[c0 + 3], alpha * hv.w);
    }
}

// ---------------------------------------------------------------------------
// X = relu(X + bias), float4-vectorized (rs % 4 == 0 for all layers).
// ---------------------------------------------------------------------------
__global__ void bias_relu_kernel(float* __restrict__ X, const float* __restrict__ bias,
                                 int Nn, int rs) {
    size_t i4 = (size_t)blockIdx.x * blockDim.x + threadIdx.x;
    size_t total4 = (size_t)Nn * rs / 4;
    if (i4 >= total4) return;
    int c = (int)((i4 * 4) % rs);
    float4 v = *(float4*)&X[i4 * 4];
    v.x += bias[c + 0]; v.y += bias[c + 1];
    v.z += bias[c + 2]; v.w += bias[c + 3];
    v.x = v.x > 0.f ? v.x : 0.f;
    v.y = v.y > 0.f ? v.y : 0.f;
    v.z = v.z > 0.f ? v.z : 0.f;
    v.w = v.w > 0.f ? v.w : 0.f;
    *(float4*)&X[i4 * 4] = v;
}

// ---------------------------------------------------------------------------
// Final: out = log_softmax(agg + bias), C = 40 classes.
// ---------------------------------------------------------------------------
__global__ void logsoftmax_kernel(const float* __restrict__ agg, const float* __restrict__ bias,
                                  float* __restrict__ out, int Nn, int C) {
    int n = blockIdx.x * blockDim.x + threadIdx.x;
    if (n >= Nn) return;
    float buf[40];
    float mx = -1e30f;
    for (int c = 0; c < C; ++c) {
        float v = agg[(size_t)n * C + c] + bias[c];
        buf[c] = v;
        mx = fmaxf(mx, v);
    }
    float ssum = 0.f;
    for (int c = 0; c < C; ++c) ssum += __expf(buf[c] - mx);
    float ls = __logf(ssum) + mx;
    for (int c = 0; c < C; ++c) out[(size_t)n * C + c] = buf[c] - ls;
}

// ---------------------------------------------------------------------------
// Host orchestration.
// ---------------------------------------------------------------------------
extern "C" void kernel_launch(void* const* d_in, const int* in_sizes, int n_in,
                              void* d_out, int out_size, void* d_ws, size_t ws_size,
                              hipStream_t stream) {
    const float* x   = (const float*)d_in[0];
    const int*   ei  = (const int*)d_in[1];
    const float* W0  = (const float*)d_in[2];
    const float* as0 = (const float*)d_in[3];
    const float* ad0 = (const float*)d_in[4];
    const float* b0  = (const float*)d_in[5];
    const float* W1  = (const float*)d_in[6];
    const float* as1 = (const float*)d_in[7];
    const float* ad1 = (const float*)d_in[8];
    const float* b1  = (const float*)d_in[9];
    const float* W2  = (const float*)d_in[10];
    const float* as2 = (const float*)d_in[11];
    const float* ad2 = (const float*)d_in[12];
    const float* b2  = (const float*)d_in[13];

    const int Nn = in_sizes[0] / 128;    // 50000
    const int E  = in_sizes[1] / 2;      // 800000
    const int EP = E + Nn;
    const int* srcA = ei;
    const int* dstA = ei + E;

    // Workspace carve-out.
    char* w = (char*)d_ws;
    auto carve = [&](size_t bytes) {
        void* p = (void*)w;
        w += (bytes + 255) & ~(size_t)255;
        return p;
    };
    float*          Hbuf  = (float*)carve((size_t)Nn * 512 * 4);   // GEMM output per layer
    float*          Fbuf  = (float*)carve((size_t)Nn * 512 * 4);   // features / aggregation
    float*          aS    = (float*)carve((size_t)Nn * 8 * 4);
    float*          aD    = (float*)carve((size_t)Nn * 8 * 4);
    unsigned*       emaxU = (unsigned*)carve((size_t)Nn * 8 * 4);
    float*          denom = (float*)carve((size_t)Nn * 8 * 4);
    float*          exw   = (float*)carve((size_t)EP * 8 * 4);
    unsigned short* Bpack = (unsigned short*)carve((size_t)512 * 512 * 2);

    auto run_layer = [&](const float* feats, int K, const float* W, int Nout,
                         int heads, int C, const float* attS, const float* attD,
                         float* Hout, float* agg) {
        const int coltiles = (Nout + 15) / 16;
        const int ksteps   = K / 32;

        // 1) Pack W into WMMA B-fragment order.
        {
            int total = coltiles * ksteps * 32;
            pack_B_kernel<<<(total + 255) / 256, 256, 0, stream>>>(
                W, Bpack, K, Nout, coltiles);
        }
        // 2) WMMA GEMM: Hout = feats @ W.  (2 col-tiles per wave, 8 waves/blk)
        {
            dim3 block(32, WPB);
            dim3 grid((coltiles + WPB * 2 - 1) / (WPB * 2), (Nn + 15) / 16);
            wmma_gemm_kernel<<<grid, block, 0, stream>>>(
                feats, Bpack, Hout, Nn, K, Nout, coltiles);
        }
        // 3) Attention dot products.
        {
            int total = Nn * heads;
            att_dots_kernel<<<(total + 255) / 256, 256, 0, stream>>>(
                Hout, attS, attD, aS, aD, Nn, heads, C, Nout);
        }
        // 4) Zero reduction targets (graph-capture-safe async memsets).
        hipMemsetAsync(emaxU, 0, (size_t)Nn * heads * 4, stream);
        hipMemsetAsync(denom, 0, (size_t)Nn * heads * 4, stream);
        hipMemsetAsync(agg,   0, (size_t)Nn * Nout * 4, stream);
        // 5) Segment softmax + message scatter.
        edge_max_kernel<<<(EP + 255) / 256, 256, 0, stream>>>(
            srcA, dstA, E, Nn, aS, aD, emaxU, heads);
        edge_expsum_kernel<<<(EP + 255) / 256, 256, 0, stream>>>(
            srcA, dstA, E, Nn, aS, aD, emaxU, exw, denom, heads);
        {
            size_t threads = (size_t)EP * 32;
            edge_scatter_kernel<<<(unsigned)((threads + 255) / 256), 256, 0, stream>>>(
                srcA, dstA, E, Nn, Hout, exw, denom, agg, heads, Nout);
        }
    };

    // ---- Layer 0: [N,128] -> 8 heads x 64, concat -> [N,512], relu ----
    run_layer(x, 128, W0, 512, 8, 64, as0, ad0, Hbuf, Fbuf);
    {
        size_t total4 = (size_t)Nn * 512 / 4;
        bias_relu_kernel<<<(unsigned)((total4 + 255) / 256), 256, 0, stream>>>(
            Fbuf, b0, Nn, 512);
    }
    // ---- Layer 1: [N,512] -> [N,512], relu (agg reuses Fbuf: dead after GEMM)
    run_layer(Fbuf, 512, W1, 512, 8, 64, as1, ad1, Hbuf, Fbuf);
    {
        size_t total4 = (size_t)Nn * 512 / 4;
        bias_relu_kernel<<<(unsigned)((total4 + 255) / 256), 256, 0, stream>>>(
            Fbuf, b1, Nn, 512);
    }
    // ---- Layer 2: [N,512] -> 1 head x 40 -> log_softmax ----
    run_layer(Fbuf, 512, W2, 40, 1, 40, as2, ad2, Hbuf, Fbuf);
    logsoftmax_kernel<<<(Nn + 255) / 256, 256, 0, stream>>>(
        Fbuf, b2, (float*)d_out, Nn, 40);
}
